// SimulatedEPMoELayer_74517682585989
// MI455X (gfx1250) — matrix operand order repaired and checked by reading, not compiled, not run on hardware
//
#include <hip/hip_runtime.h>
#include <hip/hip_bf16.h>
#include <stdint.h>

// ---------------- problem constants ----------------
#define T_TOK 4096      // B*S tokens
#define D_DIM 2048
#define F_DIM 1408
#define NEXP  16
#define NLOC  8

// ---------------- GEMM tiling ----------------
#define BM 64
#define BN 128          // block N tile (per-wave 32x32, waves 2(M) x 4(N))
#define BK 32
#define APITCH 40       // bf16 elems; 80B rows -> 16B-aligned segments
#define BPITCH 40

typedef __bf16 bf16_t;
typedef bf16_t v16bf __attribute__((ext_vector_type(16)));
typedef float  v8f   __attribute__((ext_vector_type(8)));

union V16 {                       // trivial members only (safe union)
  v16bf          v;
  unsigned short s[16];
};

// hardware conversion: lowers to v_cvt_pk_bf16_f32 / v_cvt_bf16 on gfx1250
__device__ __forceinline__ bf16_t f32_to_bf16(float f) { return (bf16_t)f; }

// ---------------- workspace layout (bytes) ----------------
// 0     : counts[8]  (int)
// 32    : fill[8]    (int)
// 64    : base[8]    (int)
// 96    : probsum[16] (float)
// 160   : usage[16]   (float)
// 256   : cw[T*8]     (float)            131072 B
// 131328: list[8*T]   (int)              131072 B
// 262656: H[2*T*F]    (bf16)             23068672 B   => total ~23.3 MB

// ================= kernel 0: zero out + header =================
__global__ void moe_zero(float* __restrict__ out, float* __restrict__ hdr, size_t n) {
  size_t i = (size_t)blockIdx.x * blockDim.x + threadIdx.x;
  size_t stride = (size_t)gridDim.x * blockDim.x;
  for (size_t j = i; j < n; j += stride) out[j] = 0.f;
  if (i < 56) hdr[i] = 0.f;   // counts/fill/base/probsum/usage (224 B)
}

// ================= kernel 1: router (one wave32 per token) =================
__global__ void moe_router(const float* __restrict__ x, const float* __restrict__ Wr,
                           float* __restrict__ cw, int* __restrict__ counts,
                           float* __restrict__ probsum, float* __restrict__ usage) {
  const int wave = threadIdx.x >> 5;
  const int lane = threadIdx.x & 31;
  const int t = blockIdx.x * (blockDim.x >> 5) + wave;
  if (t >= T_TOK) return;
  const float* xr = x + (size_t)t * D_DIM;

  float acc[NEXP];
#pragma unroll
  for (int e = 0; e < NEXP; ++e) acc[e] = 0.f;
  for (int d = lane; d < D_DIM; d += 32) {
    float xv = xr[d];
    const float* wr = Wr + (size_t)d * NEXP;
#pragma unroll
    for (int e = 0; e < NEXP; ++e) acc[e] += xv * wr[e];
  }
#pragma unroll
  for (int e = 0; e < NEXP; ++e)
    for (int m = 16; m >= 1; m >>= 1)
      acc[e] += __shfl_xor(acc[e], m, 32);

  // softmax (replicated per lane; all lanes hold full sums)
  float mx = acc[0];
#pragma unroll
  for (int e = 1; e < NEXP; ++e) mx = fmaxf(mx, acc[e]);
  float p[NEXP]; float sum = 0.f;
#pragma unroll
  for (int e = 0; e < NEXP; ++e) { p[e] = __expf(acc[e] - mx); sum += p[e]; }
  float inv = 1.f / sum;
#pragma unroll
  for (int e = 0; e < NEXP; ++e) p[e] *= inv;

  // top-2 (first index wins ties, matching jax.lax.top_k)
  int id1 = 0; float p1 = p[0];
#pragma unroll
  for (int e = 1; e < NEXP; ++e) if (p[e] > p1) { p1 = p[e]; id1 = e; }
  int id2 = -1; float p2 = -1.f;
#pragma unroll
  for (int e = 0; e < NEXP; ++e) if (e != id1 && p[e] > p2) { p2 = p[e]; id2 = e; }
  float inv2 = 1.f / (p1 + p2 + 1e-9f);
  float w1 = p1 * inv2, w2 = p2 * inv2;

  if (lane < NLOC) {
    float c = ((id1 == lane) ? w1 : 0.f) + ((id2 == lane) ? w2 : 0.f);
    cw[t * NLOC + lane] = c;
    if (c > 0.f) atomicAdd(&counts[lane], 1);
  }
  // lane-indexed select without scratch spill
  float myp = 0.f;
#pragma unroll
  for (int e = 0; e < NEXP; ++e) myp = (e == lane) ? p[e] : myp;
  if (lane < NEXP) atomicAdd(&probsum[lane], myp);
  if (lane == 0)   atomicAdd(&usage[id1], 1.0f);
}

// ================= kernel 2: prefix bases + aux loss =================
__global__ void moe_prefix_aux(const int* __restrict__ counts, int* __restrict__ base,
                               const float* __restrict__ probsum, const float* __restrict__ usage,
                               float* __restrict__ out_aux) {
  if (threadIdx.x == 0 && blockIdx.x == 0) {
    int run = 0;
    for (int e = 0; e < NLOC; ++e) { base[e] = run; run += counts[e]; }
    const float invT = 1.0f / (float)T_TOK;
    float s = 0.f;
    for (int e = 0; e < NEXP; ++e) s += (usage[e] * invT) * (probsum[e] * invT);
    *out_aux = (float)NEXP * s;
  }
}

// ================= kernel 3: compact token lists per expert =================
__global__ void moe_fill(const float* __restrict__ cw, int* __restrict__ fill,
                         int* __restrict__ list) {
  int t = blockIdx.x * blockDim.x + threadIdx.x;
  if (t >= T_TOK) return;
#pragma unroll
  for (int e = 0; e < NLOC; ++e) {
    if (cw[t * NLOC + e] > 0.f) {
      int pos = atomicAdd(&fill[e], 1);
      list[e * T_TOK + pos] = t;
    }
  }
}

// ================= kernel 4: H = silu(X@Wg) * (X@Wu), gathered rows =================
__global__ __launch_bounds__(256)
void moe_ffn_h(const float* __restrict__ x, const float* __restrict__ Wg,
               const float* __restrict__ Wu, const int* __restrict__ counts,
               const int* __restrict__ base, const int* __restrict__ list,
               bf16_t* __restrict__ H) {
  const int e  = blockIdx.z;
  const int n  = counts[e];
  const int m0 = blockIdx.y * BM;
  if (m0 >= n) return;
  const int f0 = blockIdx.x * BN;

  __shared__ __align__(16) bf16_t lA [BM][APITCH];
  __shared__ __align__(16) bf16_t lBg[BN][BPITCH];   // transposed: [col][k]
  __shared__ __align__(16) bf16_t lBu[BN][BPITCH];

  const int tid  = threadIdx.x;
  const int lane = tid & 31;
  const int wave = tid >> 5;
  const int wm   = wave >> 2;        // 0..1 : 32-row strip
  const int wn   = wave & 3;         // 0..3 : 32-col strip
  const int mrow = lane & 15;
  const int half = lane >> 4;

  // A-load: each thread loads 8 consecutive K of one row
  const int arow = tid >> 2;           // 0..63
  const int aseg = (tid & 3) * 8;      // K segment
  int tokA = -1;
  if (m0 + arow < n) tokA = list[e * T_TOK + m0 + arow];
  const float* xrow = (tokA >= 0) ? (x + (size_t)tokA * D_DIM) : x;

  // B-load: each thread loads 16 consecutive F of one K-row
  const int brow = tid >> 3;           // K row 0..31
  const int bseg = (tid & 7) * 16;     // F col segment
  const float* WgB = Wg + (size_t)e * D_DIM * F_DIM + (size_t)f0 + bseg;
  const float* WuB = Wu + (size_t)e * D_DIM * F_DIM + (size_t)f0 + bseg;

  const v8f vz = {0.f,0.f,0.f,0.f,0.f,0.f,0.f,0.f};
  v8f aG00 = vz, aG01 = vz, aG10 = vz, aG11 = vz;
  v8f aU00 = vz, aU01 = vz, aU10 = vz, aU11 = vz;

  for (int k0 = 0; k0 < D_DIM; k0 += BK) {
    // ---- A tile 64x32 fp32 -> bf16
    float4 fa = {0,0,0,0}, fb = {0,0,0,0};
    if (tokA >= 0) {
      const float4* s4 = (const float4*)(xrow + k0 + aseg);
      fa = s4[0]; fb = s4[1];
    }
    {
      bf16_t* d = &lA[arow][aseg];
      d[0]=f32_to_bf16(fa.x); d[1]=f32_to_bf16(fa.y);
      d[2]=f32_to_bf16(fa.z); d[3]=f32_to_bf16(fa.w);
      d[4]=f32_to_bf16(fb.x); d[5]=f32_to_bf16(fb.y);
      d[6]=f32_to_bf16(fb.z); d[7]=f32_to_bf16(fb.w);
    }
    // ---- B tiles 32x128 fp32 -> bf16, stored [col][k]
    {
      const float4* sg = (const float4*)(WgB + (size_t)(k0 + brow) * F_DIM);
      const float4* su = (const float4*)(WuB + (size_t)(k0 + brow) * F_DIM);
#pragma unroll
      for (int q = 0; q < 4; ++q) {
        float4 g = sg[q], u = su[q];
        lBg[bseg + q*4 + 0][brow] = f32_to_bf16(g.x);
        lBg[bseg + q*4 + 1][brow] = f32_to_bf16(g.y);
        lBg[bseg + q*4 + 2][brow] = f32_to_bf16(g.z);
        lBg[bseg + q*4 + 3][brow] = f32_to_bf16(g.w);
        lBu[bseg + q*4 + 0][brow] = f32_to_bf16(u.x);
        lBu[bseg + q*4 + 1][brow] = f32_to_bf16(u.y);
        lBu[bseg + q*4 + 2][brow] = f32_to_bf16(u.z);
        lBu[bseg + q*4 + 3][brow] = f32_to_bf16(u.w);
      }
    }
    __syncthreads();
    if (k0 + BK < D_DIM) {               // gfx1250 global_prefetch_b8
      __builtin_prefetch(WgB + (size_t)(k0 + BK + brow) * F_DIM, 0, 1);
      __builtin_prefetch(WuB + (size_t)(k0 + BK + brow) * F_DIM, 0, 1);
    }
    // ---- fragments in contiguous k order (A: lane half h holds K {16h..16h+15};
    //      B: lane half h holds K {16h..16h+15}); each as two 16B ds loads.
    V16 a0, a1, bg0, bg1, bu0, bu1;
    {
      const bf16_t* pr0 = &lA[wm*32      + mrow][0];
      const bf16_t* pr1 = &lA[wm*32 + 16 + mrow][0];
      *(uint4*)&a0.s[0] = *(const uint4*)(pr0 +     16*half);
      *(uint4*)&a0.s[8] = *(const uint4*)(pr0 + 8 + 16*half);
      *(uint4*)&a1.s[0] = *(const uint4*)(pr1 +     16*half);
      *(uint4*)&a1.s[8] = *(const uint4*)(pr1 + 8 + 16*half);
      const bf16_t* pg0 = &lBg[wn*32      + mrow][16*half];
      const bf16_t* pg1 = &lBg[wn*32 + 16 + mrow][16*half];
      const bf16_t* pu0 = &lBu[wn*32      + mrow][16*half];
      const bf16_t* pu1 = &lBu[wn*32 + 16 + mrow][16*half];
      *(uint4*)&bg0.s[0] = *(const uint4*)(pg0);
      *(uint4*)&bg0.s[8] = *(const uint4*)(pg0 + 8);
      *(uint4*)&bg1.s[0] = *(const uint4*)(pg1);
      *(uint4*)&bg1.s[8] = *(const uint4*)(pg1 + 8);
      *(uint4*)&bu0.s[0] = *(const uint4*)(pu0);
      *(uint4*)&bu0.s[8] = *(const uint4*)(pu0 + 8);
      *(uint4*)&bu1.s[0] = *(const uint4*)(pu1);
      *(uint4*)&bu1.s[8] = *(const uint4*)(pu1 + 8);
    }
    aG00 = __builtin_amdgcn_wmma_f32_16x16x32_bf16(false, a0.v, false, bg0.v, (short)0, aG00, false, false);
    aG01 = __builtin_amdgcn_wmma_f32_16x16x32_bf16(false, a0.v, false, bg1.v, (short)0, aG01, false, false);
    aG10 = __builtin_amdgcn_wmma_f32_16x16x32_bf16(false, a1.v, false, bg0.v, (short)0, aG10, false, false);
    aG11 = __builtin_amdgcn_wmma_f32_16x16x32_bf16(false, a1.v, false, bg1.v, (short)0, aG11, false, false);
    aU00 = __builtin_amdgcn_wmma_f32_16x16x32_bf16(false, a0.v, false, bu0.v, (short)0, aU00, false, false);
    aU01 = __builtin_amdgcn_wmma_f32_16x16x32_bf16(false, a0.v, false, bu1.v, (short)0, aU01, false, false);
    aU10 = __builtin_amdgcn_wmma_f32_16x16x32_bf16(false, a1.v, false, bu0.v, (short)0, aU10, false, false);
    aU11 = __builtin_amdgcn_wmma_f32_16x16x32_bf16(false, a1.v, false, bu1.v, (short)0, aU11, false, false);
    __syncthreads();
  }

  const int baseE = base[e];
  const int colg0 = f0 + wn*32      + mrow;
  const int colg1 = f0 + wn*32 + 16 + mrow;
#pragma unroll
  for (int r = 0; r < 8; ++r) {
    int lr0 = wm*32 + r + 8*half;
    int lr1 = lr0 + 16;
    if (m0 + lr0 < n) {
      size_t rowoff = (size_t)(baseE + m0 + lr0) * F_DIM;
      float g0 = aG00[r], u0 = aU00[r];
      H[rowoff + colg0] = f32_to_bf16((g0 / (1.f + __expf(-g0))) * u0);
      float g1 = aG01[r], u1 = aU01[r];
      H[rowoff + colg1] = f32_to_bf16((g1 / (1.f + __expf(-g1))) * u1);
    }
    if (m0 + lr1 < n) {
      size_t rowoff = (size_t)(baseE + m0 + lr1) * F_DIM;
      float g0 = aG10[r], u0 = aU10[r];
      H[rowoff + colg0] = f32_to_bf16((g0 / (1.f + __expf(-g0))) * u0);
      float g1 = aG11[r], u1 = aU11[r];
      H[rowoff + colg1] = f32_to_bf16((g1 / (1.f + __expf(-g1))) * u1);
    }
  }
}

// ================= kernel 5: OUT += (H @ Wd) * cw, scatter rows =================
__global__ __launch_bounds__(256)
void moe_ffn_out(const bf16_t* __restrict__ H, const float* __restrict__ Wd,
                 const int* __restrict__ counts, const int* __restrict__ base,
                 const int* __restrict__ list, const float* __restrict__ cw,
                 float* __restrict__ out) {
  const int e  = blockIdx.z;
  const int n  = counts[e];
  const int m0 = blockIdx.y * BM;
  if (m0 >= n) return;
  const int d0 = blockIdx.x * BN;

  __shared__ __align__(16) bf16_t lA[BM][APITCH];
  __shared__ __align__(16) bf16_t lB[BN][BPITCH];   // transposed: [col][k]

  const int tid  = threadIdx.x;
  const int lane = tid & 31;
  const int wave = tid >> 5;
  const int wm   = wave >> 2;
  const int wn   = wave & 3;
  const int mrow = lane & 15;
  const int half = lane >> 4;

  const int arow = tid >> 2;
  const int aseg = (tid & 3) * 8;
  const bool avalid = (m0 + arow < n);
  const bf16_t* hrow = H + (size_t)(base[e] + m0 + arow) * F_DIM;

  const int brow = tid >> 3;           // K row 0..31
  const int bseg = (tid & 7) * 16;     // D col segment
  const float* WdB = Wd + (size_t)e * F_DIM * D_DIM + (size_t)d0 + bseg;

  const v8f vz = {0.f,0.f,0.f,0.f,0.f,0.f,0.f,0.f};
  v8f a00 = vz, a01 = vz, a10 = vz, a11 = vz;

  for (int k0 = 0; k0 < F_DIM; k0 += BK) {
    uint4 av = make_uint4(0u, 0u, 0u, 0u);
    if (avalid) av = *(const uint4*)(hrow + k0 + aseg);
    *(uint4*)&lA[arow][aseg] = av;
    {
      const float4* sd = (const float4*)(WdB + (size_t)(k0 + brow) * D_DIM);
#pragma unroll
      for (int q = 0; q < 4; ++q) {
        float4 b = sd[q];
        lB[bseg + q*4 + 0][brow] = f32_to_bf16(b.x);
        lB[bseg + q*4 + 1][brow] = f32_to_bf16(b.y);
        lB[bseg + q*4 + 2][brow] = f32_to_bf16(b.z);
        lB[bseg + q*4 + 3][brow] = f32_to_bf16(b.w);
      }
    }
    __syncthreads();
    if (k0 + BK < F_DIM)
      __builtin_prefetch(WdB + (size_t)(k0 + BK + brow) * D_DIM, 0, 1);

    V16 a0, a1, b0, b1;
    {
      const bf16_t* pr0 = &lA[wm*32      + mrow][0];
      const bf16_t* pr1 = &lA[wm*32 + 16 + mrow][0];
      *(uint4*)&a0.s[0] = *(const uint4*)(pr0 +     16*half);
      *(uint4*)&a0.s[8] = *(const uint4*)(pr0 + 8 + 16*half);
      *(uint4*)&a1.s[0] = *(const uint4*)(pr1 +     16*half);
      *(uint4*)&a1.s[8] = *(const uint4*)(pr1 + 8 + 16*half);
      const bf16_t* pb0 = &lB[wn*32      + mrow][16*half];
      const bf16_t* pb1 = &lB[wn*32 + 16 + mrow][16*half];
      *(uint4*)&b0.s[0] = *(const uint4*)(pb0);
      *(uint4*)&b0.s[8] = *(const uint4*)(pb0 + 8);
      *(uint4*)&b1.s[0] = *(const uint4*)(pb1);
      *(uint4*)&b1.s[8] = *(const uint4*)(pb1 + 8);
    }
    a00 = __builtin_amdgcn_wmma_f32_16x16x32_bf16(false, a0.v, false, b0.v, (short)0, a00, false, false);
    a01 = __builtin_amdgcn_wmma_f32_16x16x32_bf16(false, a0.v, false, b1.v, (short)0, a01, false, false);
    a10 = __builtin_amdgcn_wmma_f32_16x16x32_bf16(false, a1.v, false, b0.v, (short)0, a10, false, false);
    a11 = __builtin_amdgcn_wmma_f32_16x16x32_bf16(false, a1.v, false, b1.v, (short)0, a11, false, false);
    __syncthreads();
  }

  const int colg0 = d0 + wn*32      + mrow;
  const int colg1 = d0 + wn*32 + 16 + mrow;
#pragma unroll
  for (int r = 0; r < 8; ++r) {
    int lr0 = wm*32 + r + 8*half;
    int lr1 = lr0 + 16;
    if (m0 + lr0 < n) {
      int tok = list[e * T_TOK + m0 + lr0];
      float c = cw[tok * NLOC + e];
      float* orow = out + (size_t)tok * D_DIM;
      atomicAdd(orow + colg0, a00[r] * c);
      atomicAdd(orow + colg1, a01[r] * c);
    }
    if (m0 + lr1 < n) {
      int tok = list[e * T_TOK + m0 + lr1];
      float c = cw[tok * NLOC + e];
      float* orow = out + (size_t)tok * D_DIM;
      atomicAdd(orow + colg0, a10[r] * c);
      atomicAdd(orow + colg1, a11[r] * c);
    }
  }
}

// ================= launcher =================
extern "C" void kernel_launch(void* const* d_in, const int* in_sizes, int n_in,
                              void* d_out, int out_size, void* d_ws, size_t ws_size,
                              hipStream_t stream) {
  (void)in_sizes; (void)n_in; (void)out_size; (void)ws_size;
  const float* x  = (const float*)d_in[0];
  const float* Wr = (const float*)d_in[1];
  const float* Wg = (const float*)d_in[2];
  const float* Wu = (const float*)d_in[3];
  const float* Wd = (const float*)d_in[4];
  float* out = (float*)d_out;

  char* ws = (char*)d_ws;
  int*   counts  = (int*)  (ws + 0);
  int*   fill    = (int*)  (ws + 32);
  int*   base    = (int*)  (ws + 64);
  float* probsum = (float*)(ws + 96);
  float* usage   = (float*)(ws + 160);
  float* cw      = (float*)(ws + 256);
  int*   list    = (int*)  (ws + 131328);
  bf16_t* H      = (bf16_t*)(ws + 262656);   // 2*T*F bf16 (~23 MB)

  const size_t outN = (size_t)T_TOK * D_DIM;   // aux scalar lives at out[outN]

  moe_zero<<<2048, 256, 0, stream>>>(out, (float*)ws, outN);
  moe_router<<<T_TOK / 4, 128, 0, stream>>>(x, Wr, cw, counts, probsum, usage);
  moe_prefix_aux<<<1, 32, 0, stream>>>(counts, base, probsum, usage, out + outN);
  moe_fill<<<T_TOK / 256, 256, 0, stream>>>(cw, fill, list);

  dim3 g1(F_DIM / BN, T_TOK / BM, NLOC);   // (11, 64, 8)
  moe_ffn_h<<<g1, 256, 0, stream>>>(x, Wg, Wu, counts, base, list, H);

  dim3 g2(D_DIM / BN, T_TOK / BM, NLOC);   // (16, 64, 8)
  moe_ffn_out<<<g2, 256, 0, stream>>>(H, Wd, counts, base, list, cw, out);
}
